// UniqGCN_9723805958219
// MI455X (gfx1250) — matrix operand order repaired and checked by reading, not compiled, not run on hardware
//
#include <hip/hip_runtime.h>

#define NODES 50000
#define EDGES 800000
#define FIN   512
#define HID   256
#define NCLS  64

typedef __attribute__((ext_vector_type(16))) __bf16 v16bf;
typedef __attribute__((ext_vector_type(8)))  float  v8f;
typedef __attribute__((ext_vector_type(4)))  unsigned v4u;
typedef __attribute__((ext_vector_type(8)))  int      v8i;
typedef __attribute__((ext_vector_type(4)))  int      v4i;

// ---------------------------------------------------------------- utilities
__device__ __forceinline__ unsigned short f2bf1(float a) {
  union { float f; unsigned u; } x; x.f = a;
  return (unsigned short)((x.u + 0x7FFFu + ((x.u >> 16) & 1u)) >> 16);  // RNE
}
__device__ __forceinline__ unsigned pack_bf2(float a, float b) {
  return (unsigned)f2bf1(a) | ((unsigned)f2bf1(b) << 16);
}
__device__ __forceinline__ float sigm(float x) { return 1.0f / (1.0f + __expf(-x)); }

// -------------------------------------------------- TDM 2D tile -> LDS
// Copies a (128 x 32) bf16 tile (row stride = strideElems) into LDS with a
// 1-dword pad after every 16 dwords (row stride 17 dwords, conflict-free).
// Rows >= rowsAvail are zero-filled by TDM OOB handling. Tracked by TENSORcnt.
__device__ __forceinline__ void tdm_load_2d(
    unsigned ldsAddr, const void* gptr, unsigned rowsAvail, unsigned strideElems)
{
  unsigned long long ga = (unsigned long long)(uintptr_t)gptr;
  v4u g0;
  g0.x = 1u;                                            // count=1, user mode
  g0.y = ldsAddr;                                       // lds_addr (bytes)
  g0.z = (unsigned)(ga & 0xFFFFFFFFu);                  // global_addr[31:0]
  g0.w = (unsigned)((ga >> 32) & 0x1FFFFFFu) | (2u << 30);  // addr[56:32]|type=2
  v8i g1;
  g1[0] = (1 << 16)      // data_size = 2 bytes
        | (1 << 20)      // pad_enable
        | (3 << 22)      // pad_interval: 16 dwords
        | (0 << 25);     // pad_amount: 1 dword
  g1[1] = (int)((32u & 0xFFFFu) << 16);                       // tensor_dim0 lo16
  g1[2] = (int)((rowsAvail & 0xFFFFu) << 16);                 // dim0 hi | dim1 lo
  g1[3] = (int)((rowsAvail >> 16) | (32u << 16));             // dim1 hi | tile_dim0
  g1[4] = 128;                                                // tile_dim1 (tile_dim2=0)
  g1[5] = (int)strideElems;                                   // dim0_stride lo32
  g1[6] = 0;                                                  // stride hi | dim1_stride lo
  g1[7] = 0;
  v4i z = { 0, 0, 0, 0 };
#if __clang_major__ >= 23
  v8i z8 = { 0, 0, 0, 0, 0, 0, 0, 0 };
  __builtin_amdgcn_tensor_load_to_lds(g0, g1, z, z, z8, 0);
#else
  __builtin_amdgcn_tensor_load_to_lds(g0, g1, z, z, 0);
#endif
}

// ------------------------------------------------- WMMA GEMM: C = A @ B^T
// A: (M x K) bf16 row-major.  B: (Ncols x K) bf16 row-major (Linear weight).
// Macro tile 128x128, K chunk 32, 8 waves, wave = 32x64 (2x4 wmma).
// LDS tiles double-buffered and filled by wave-0-issued TDM transfers.
__device__ __forceinline__ void mm_stage(
    const unsigned short* __restrict__ A, const unsigned short* __restrict__ B,
    int K, int M, int Ncols, int rowBase, int colBase,
    unsigned* lA0, unsigned* lB0, unsigned* lA1, unsigned* lB1,
    v8f (&acc)[2][4])
{
  const int tid  = threadIdx.x;
  const int lane = tid & 31;
  const int wave = tid >> 5;
  const int wm   = wave & 3;     // wave row (4 along M)
  const int wn   = wave >> 2;    // wave col (2 along N, 64 cols each)
  const int half = lane >> 4;
  const int l15  = lane & 15;
  const unsigned aRows = (unsigned)(M - rowBase);
  const unsigned bRows = (unsigned)(Ncols - colBase);

  auto issue = [&](int tile, unsigned* dA, unsigned* dB) {
    int kk = tile * 32;
    tdm_load_2d((unsigned)(uintptr_t)dA, A + (size_t)rowBase * K + kk, aRows, (unsigned)K);
    tdm_load_2d((unsigned)(uintptr_t)dB, B + (size_t)colBase * K + kk, bRows, (unsigned)K);
  };
  auto compute = [&](const unsigned* sA, const unsigned* sB) {
    union Frag { unsigned u[8]; v16bf v; };
    Frag af[2];
    #pragma unroll
    for (int mt = 0; mt < 2; ++mt) {
      int r = wm * 32 + mt * 16 + l15;
      #pragma unroll
      for (int vv = 0; vv < 8; ++vv) {
        // ISA A layout: K pair base = (v>>2)*16 + half*8 + (v&3)*2
        int kc = ((vv >> 2) << 3) + (half << 2) + (vv & 3);
        af[mt].u[vv] = sA[r * 17 + kc];
      }
    }
    #pragma unroll
    for (int nt = 0; nt < 4; ++nt) {
      int n = wn * 64 + nt * 16 + l15;
      Frag bf;
      #pragma unroll
      for (int vv = 0; vv < 8; ++vv) {
        // ISA B layout: K pair base = (v>>2)*16 + half*8 + (v&3)*2
        int kc = ((vv >> 2) << 3) + (half << 2) + (vv & 3);
        bf.u[vv] = sB[n * 17 + kc];
      }
      #pragma unroll
      for (int mt = 0; mt < 2; ++mt)
        acc[mt][nt] = __builtin_amdgcn_wmma_f32_16x16x32_bf16(
            false, af[mt].v, false, bf.v, (short)0, acc[mt][nt], false, false);
    }
  };

  const int nk = K / 32;
  __syncthreads();                       // buffers free of prior readers
  if (wave == 0) {
    issue(0, lA0, lB0);
    if (nk > 1) issue(1, lA1, lB1);      // keep two tiles in flight
  }
  for (int i = 0; i < nk; ++i) {
    if (wave == 0) {
      if (i + 1 < nk) __builtin_amdgcn_s_wait_tensorcnt((short)2);  // tile i done
      else            __builtin_amdgcn_s_wait_tensorcnt((short)0);
    }
    __syncthreads();                     // tile i resident for all waves
    if (i & 1) compute(lA1, lB1); else compute(lA0, lB0);
    if (i + 2 < nk) {
      __syncthreads();                   // buffer fully consumed
      if (wave == 0) issue(i + 2, (i & 1) ? lA1 : lA0, (i & 1) ? lB1 : lB0);
    }
  }
}

__global__ void __launch_bounds__(256) k_gemm(
    const unsigned short* __restrict__ A1, const unsigned short* __restrict__ B1, int K1,
    const unsigned short* __restrict__ A2, const unsigned short* __restrict__ B2, int K2,
    const float* __restrict__ bias1, const float* __restrict__ bias2,
    float* __restrict__ C, int M, int Ncols, int relu)
{
  __shared__ unsigned lds[4 * 128 * 17];
  unsigned* lA0 = lds;
  unsigned* lB0 = lds + 128 * 17;
  unsigned* lA1 = lds + 2 * 128 * 17;
  unsigned* lB1 = lds + 3 * 128 * 17;

  v8f acc[2][4];
  #pragma unroll
  for (int a = 0; a < 2; ++a)
    #pragma unroll
    for (int b = 0; b < 4; ++b)
      acc[a][b] = (v8f)0.0f;

  const int rowBase = blockIdx.y * 128;
  const int colBase = blockIdx.x * 128;

  mm_stage(A1, B1, K1, M, Ncols, rowBase, colBase, lA0, lB0, lA1, lB1, acc);
  if (A2) mm_stage(A2, B2, K2, M, Ncols, rowBase, colBase, lA0, lB0, lA1, lB1, acc);

  const int lane = threadIdx.x & 31;
  const int wave = threadIdx.x >> 5;
  const int wm = wave & 3, wn = wave >> 2;
  const int half = lane >> 4, l15 = lane & 15;
  #pragma unroll
  for (int mt = 0; mt < 2; ++mt)
    #pragma unroll
    for (int nt = 0; nt < 4; ++nt)
      #pragma unroll
      for (int d = 0; d < 8; ++d) {
        int m = rowBase + wm * 32 + mt * 16 + half * 8 + d;  // C/D: m = d + 8*(lane>=16)
        int n = colBase + wn * 64 + nt * 16 + l15;
        if (m < M && n < Ncols) {
          float v = acc[mt][nt][d];
          if (bias1) v += bias1[n];
          if (bias2) v += bias2[n];
          if (relu)  v = fmaxf(v, 0.0f);
          C[(size_t)m * Ncols + n] = v;
        }
      }
}

// ---------------------------------------------------------- small kernels
__global__ void __launch_bounds__(256) k_fill(float* p, size_t n, float v) {
  size_t i = (size_t)blockIdx.x * 256 + threadIdx.x;
  if (i < n) p[i] = v;
}
__global__ void __launch_bounds__(256) k_f2bf(
    const float* __restrict__ s, unsigned* __restrict__ d, size_t npairs) {
  size_t i = (size_t)blockIdx.x * 256 + threadIdx.x;
  if (i < npairs) d[i] = pack_bf2(s[2 * i], s[2 * i + 1]);
}
__global__ void __launch_bounds__(256) k_transpose_bf(
    const float* __restrict__ s, unsigned short* __restrict__ d, int R, int C) {
  size_t i = (size_t)blockIdx.x * 256 + threadIdx.x;
  if (i >= (size_t)R * C) return;
  int r = (int)(i / C), c = (int)(i % C);
  d[(size_t)c * R + r] = f2bf1(s[i]);   // d is (C x R) bf16 = s^T
}
__global__ void __launch_bounds__(256) k_deg(const int* __restrict__ col, float* deg, int e) {
  int i = blockIdx.x * 256 + threadIdx.x;
  if (i < e) atomicAdd(&deg[col[i]], 1.0f);
}
__global__ void __launch_bounds__(256) k_rsqrt(float* d, int n) {
  int i = blockIdx.x * 256 + threadIdx.x;
  if (i < n) d[i] = rsqrtf(fmaxf(d[i], 1.0f));
}
// edge scatter: agg[col] += dinv[row]*dinv[col] * hw[row];  + N self loops
__global__ void __launch_bounds__(256) k_scatter(
    const int* __restrict__ row, const int* __restrict__ col,
    const float* __restrict__ dinv, const float* __restrict__ hw,
    float* __restrict__ agg, int e, int n, int C)
{
  size_t i = (size_t)blockIdx.x * 256 + threadIdx.x;
  size_t tot = (size_t)(e + n) * C;
  if (i >= tot) return;
  int eid = (int)(i / C), ch = (int)(i % C);
  int r, c;
  if (eid < e) { r = row[eid]; c = col[eid]; } else { r = c = eid - e; }
  float w = dinv[r] * dinv[c];
  atomicAdd(&agg[(size_t)c * C + ch], w * hw[(size_t)r * C + ch]);
}
// out = (relu)(agg + b), plus optional bf16 mirror for later WMMA consumers
__global__ void __launch_bounds__(256) k_bias_act(
    const float* __restrict__ agg, const float* __restrict__ b,
    float* __restrict__ out, unsigned short* __restrict__ out_bf,
    int n, int C, int relu)
{
  size_t i = (size_t)blockIdx.x * 256 + threadIdx.x;
  if (i >= (size_t)n * C) return;
  float v = agg[i] + b[i % C];
  if (relu) v = fmaxf(v, 0.0f);
  out[i] = v;
  if (out_bf) out_bf[i] = f2bf1(v);
}
// LSTM cell (gate order i,f,g,o); writes f32 h (for scores) + bf16 h (for GEMM)
__global__ void __launch_bounds__(256) k_cell(
    const float* __restrict__ g, float* __restrict__ h,
    unsigned short* __restrict__ hbf, float* __restrict__ c, int n, int H)
{
  size_t i = (size_t)blockIdx.x * 256 + threadIdx.x;
  if (i >= (size_t)n * H) return;
  int node = (int)(i / H), j = (int)(i % H);
  const float* gr = g + (size_t)node * 4 * H;
  float ig = sigm(gr[j]);
  float fg = sigm(gr[H + j]);
  float gg = tanhf(gr[2 * H + j]);
  float og = sigm(gr[3 * H + j]);
  float cn = fg * c[i] + ig * gg;
  c[i] = cn;
  float hv = og * tanhf(cn);
  h[i] = hv;
  hbf[i] = f2bf1(hv);
}
// one wave per node: score[n] (+)= dot(h[n,:], attw[off:off+H])
__global__ void __launch_bounds__(256) k_score(
    const float* __restrict__ h, const float* __restrict__ attw, int off,
    float* __restrict__ score, int n, int H, int accum)
{
  int w = (blockIdx.x * 256 + threadIdx.x) >> 5;
  int lane = threadIdx.x & 31;
  if (w >= n) return;
  const float* hr = h + (size_t)w * H;
  float s = 0.0f;
  for (int j = lane; j < H; j += 32) s += hr[j] * attw[off + j];
  #pragma unroll
  for (int o = 16; o > 0; o >>= 1) s += __shfl_down(s, o, 32);
  if (lane == 0) { if (accum) score[w] += s; else score[w] = s; }
}
__global__ void __launch_bounds__(256) k_alpha(
    const float* __restrict__ score, const float* __restrict__ attb,
    float* __restrict__ alpha, int n, int L)
{
  int i = blockIdx.x * 256 + threadIdx.x;
  if (i >= n) return;
  float b = attb[0];
  float s[5]; float mx = -1e30f;
  for (int l = 0; l < L; ++l) { s[l] = score[(size_t)l * n + i] + b; mx = fmaxf(mx, s[l]); }
  float sum = 0.0f;
  for (int l = 0; l < L; ++l) { s[l] = __expf(s[l] - mx); sum += s[l]; }
  float inv = 1.0f / sum;
  for (int l = 0; l < L; ++l) alpha[(size_t)l * n + i] = s[l] * inv;
}
__global__ void __launch_bounds__(256) k_combine(
    const float* x0, const float* x1, const float* x2, const float* x3, const float* x4,
    const float* __restrict__ alpha, float* __restrict__ out,
    unsigned short* __restrict__ out_bf, int n, int C, int L)
{
  size_t i = (size_t)blockIdx.x * 256 + threadIdx.x;
  if (i >= (size_t)n * C) return;
  int node = (int)(i / C);
  const float* xs[5] = { x0, x1, x2, x3, x4 };
  float a = 0.0f;
  for (int l = 0; l < L; ++l) a += alpha[(size_t)l * n + node] * xs[l][i];
  out[i] = a;
  out_bf[i] = f2bf1(a);
}

// ------------------------------------------------------------------- host
extern "C" void kernel_launch(void* const* d_in, const int* in_sizes, int n_in,
                              void* d_out, int out_size, void* d_ws, size_t ws_size,
                              hipStream_t stream) {
  (void)in_sizes; (void)n_in; (void)out_size; (void)ws_size;
  const float* x    = (const float*)d_in[0];
  const int*   edge = (const int*)d_in[1];
  const int*   row  = edge;          // edge_index[0] = source
  const int*   colA = edge + EDGES;  // edge_index[1] = target

  int p = 2;
  auto F = [&](int i) { return (const float*)d_in[i]; };
  const float* inW = F(p++); const float* inB = F(p++);
  const float* layW[4]; const float* layB[4];
  for (int i = 0; i < 4; ++i) { layW[i] = F(p++); layB[i] = F(p++); }
  struct JKp { const float *Wih_f,*Whh_f,*bih_f,*bhh_f,*Wih_b,*Whh_b,*bih_b,*bhh_b,*attw,*attb; };
  JKp jk[5];  // jks[0..3] then out_jk
  for (int i = 0; i < 5; ++i) {
    jk[i].Wih_f = F(p++); jk[i].Whh_f = F(p++); jk[i].bih_f = F(p++); jk[i].bhh_f = F(p++);
    jk[i].Wih_b = F(p++); jk[i].Whh_b = F(p++); jk[i].bih_b = F(p++); jk[i].bhh_b = F(p++);
    jk[i].attw  = F(p++); jk[i].attb  = F(p++);
  }
  const float* outW = F(p++); const float* outB = F(p++);

  // workspace carve
  char* wsp = (char*)d_ws;
  auto carve = [&](size_t bytes) -> void* {
    void* r = (void*)wsp; wsp += (bytes + 255) & ~(size_t)255; return r;
  };
  float* dinv  = (float*)carve((size_t)NODES * 4);
  float* xs[5];           unsigned short* xsb[5];
  for (int i = 0; i < 5; ++i) xs[i]  = (float*)carve((size_t)NODES * HID * 4);
  for (int i = 0; i < 5; ++i) xsb[i] = (unsigned short*)carve((size_t)NODES * HID * 2);
  float* hw    = (float*)carve((size_t)NODES * HID * 4);
  float* agg   = (float*)carve((size_t)NODES * HID * 4);
  float* hbuf  = (float*)carve((size_t)NODES * 640 * 4);
  unsigned short* hbf = (unsigned short*)carve((size_t)NODES * 640 * 2);
  float* cbuf  = (float*)carve((size_t)NODES * 640 * 4);
  float* gates = (float*)carve((size_t)NODES * 2560 * 4);
  float* score = (float*)carve((size_t)5 * NODES * 4);
  float* alpha = (float*)carve((size_t)5 * NODES * 4);
  float* jkout = (float*)carve((size_t)NODES * HID * 4);
  unsigned short* jkoutb = (unsigned short*)carve((size_t)NODES * HID * 2);
  unsigned short* xb     = (unsigned short*)carve((size_t)NODES * FIN * 2);
  unsigned short* inWT   = (unsigned short*)carve((size_t)FIN * HID * 2);
  unsigned short* layWT[4];
  for (int i = 0; i < 4; ++i) layWT[i] = (unsigned short*)carve((size_t)HID * HID * 2);
  unsigned short* outWT  = (unsigned short*)carve((size_t)HID * NCLS * 2);
  unsigned short *WihF[5], *WhhF[5], *WihB[5], *WhhB[5];
  for (int i = 0; i < 5; ++i) {
    int H = 128 * (i + 1);
    WihF[i] = (unsigned short*)carve((size_t)4 * H * HID * 2);
    WhhF[i] = (unsigned short*)carve((size_t)4 * H * H * 2);
    WihB[i] = (unsigned short*)carve((size_t)4 * H * HID * 2);
    WhhB[i] = (unsigned short*)carve((size_t)4 * H * H * 2);
  }

  auto nb = [](size_t n) { return (unsigned)((n + 255) / 256); };
  auto fill = [&](float* ptr, size_t n, float v) {
    k_fill<<<nb(n), 256, 0, stream>>>(ptr, n, v);
  };
  auto cvt = [&](const float* s, unsigned short* d, size_t n) {  // n even
    k_f2bf<<<nb(n / 2), 256, 0, stream>>>(s, (unsigned*)d, n / 2);
  };
  auto gemm = [&](const unsigned short* A1, const unsigned short* B1, int K1,
                  const unsigned short* A2, const unsigned short* B2, int K2,
                  const float* bi1, const float* bi2,
                  float* C, int M, int Nc, int relu) {
    dim3 g((unsigned)((Nc + 127) / 128), (unsigned)((M + 127) / 128));
    k_gemm<<<g, 256, 0, stream>>>(A1, B1, K1, A2, B2, K2, bi1, bi2, C, M, Nc, relu);
  };
  auto conv = [&](const unsigned short* inb, const unsigned short* WT, int K, int Cout,
                  const float* b, float* out, unsigned short* outb, int relu) {
    gemm(inb, WT, K, nullptr, nullptr, 0, nullptr, nullptr, hw, NODES, Cout, 0);
    fill(agg, (size_t)NODES * Cout, 0.0f);
    size_t tot = (size_t)(EDGES + NODES) * Cout;
    k_scatter<<<nb(tot), 256, 0, stream>>>(row, colA, dinv, hw, agg, EDGES, NODES, Cout);
    k_bias_act<<<nb((size_t)NODES * Cout), 256, 0, stream>>>(agg, b, out, outb, NODES, Cout, relu);
  };
  auto jkrun = [&](int idx, int L, const JKp& J) {
    int H = 128 * L;
    for (int dir = 0; dir < 2; ++dir) {
      const unsigned short* Wih = dir ? WihB[idx] : WihF[idx];
      const unsigned short* Whh = dir ? WhhB[idx] : WhhF[idx];
      const float* bih = dir ? J.bih_b : J.bih_f;
      const float* bhh = dir ? J.bhh_b : J.bhh_f;
      fill((float*)hbf, (size_t)NODES * H / 2, 0.0f);   // bf16 zeros
      fill(cbuf, (size_t)NODES * H, 0.0f);
      for (int t = 0; t < L; ++t) {
        int l = dir ? (L - 1 - t) : t;   // logical layer processed this step
        // gates = xs[l] @ Wih^T + h @ Whh^T + bih + bhh   (fused two-stage WMMA)
        gemm(xsb[l], Wih, HID, hbf, Whh, H, bih, bhh, gates, NODES, 4 * H, 0);
        k_cell<<<nb((size_t)NODES * H), 256, 0, stream>>>(gates, hbuf, hbf, cbuf, NODES, H);
        k_score<<<nb((size_t)NODES * 32), 256, 0, stream>>>(
            hbuf, J.attw, dir * H, score + (size_t)l * NODES, NODES, H, dir);
      }
    }
    k_alpha<<<nb(NODES), 256, 0, stream>>>(score, J.attb, alpha, NODES, L);
    k_combine<<<nb((size_t)NODES * HID), 256, 0, stream>>>(
        xs[0], xs[1], xs[2], xs[3], xs[4], alpha, jkout, jkoutb, NODES, HID, L);
  };

  // symmetric GCN norm: deg = 1 (self loop) + in-degree; dinv = rsqrt(deg)
  fill(dinv, NODES, 1.0f);
  k_deg<<<nb(EDGES), 256, 0, stream>>>(colA, dinv, EDGES);
  k_rsqrt<<<nb(NODES), 256, 0, stream>>>(dinv, NODES);

  // one-time conversions: features, transposed conv weights, LSTM weights -> bf16
  cvt(x, xb, (size_t)NODES * FIN);
  k_transpose_bf<<<nb((size_t)FIN * HID), 256, 0, stream>>>(inW, inWT, FIN, HID);
  for (int i = 0; i < 4; ++i)
    k_transpose_bf<<<nb((size_t)HID * HID), 256, 0, stream>>>(layW[i], layWT[i], HID, HID);
  k_transpose_bf<<<nb((size_t)HID * NCLS), 256, 0, stream>>>(outW, outWT, HID, NCLS);
  for (int i = 0; i < 5; ++i) {
    int H = 128 * (i + 1);
    cvt(jk[i].Wih_f, WihF[i], (size_t)4 * H * HID);
    cvt(jk[i].Whh_f, WhhF[i], (size_t)4 * H * H);
    cvt(jk[i].Wih_b, WihB[i], (size_t)4 * H * HID);
    cvt(jk[i].Whh_b, WhhB[i], (size_t)4 * H * H);
  }

  conv(xb, inWT, FIN, HID, inB, xs[0], xsb[0], 1);
  for (int i = 0; i < 4; ++i) {
    jkrun(i, i + 1, jk[i]);
    conv(jkoutb, layWT[i], HID, HID, layB[i], xs[i + 1], xsb[i + 1], 1);
  }
  jkrun(4, 5, jk[4]);
  conv(jkoutb, outWT, HID, NCLS, outB, (float*)d_out, nullptr, 0);
}
